// EdgePredictionGNNModel_27779848471357
// MI455X (gfx1250) — matrix operand order repaired and checked by reading, not compiled, not run on hardware
//
#include <hip/hip_runtime.h>
#include <hip/hip_bf16.h>

typedef unsigned short u16;
typedef __attribute__((ext_vector_type(16))) __bf16 v16bf;
typedef __attribute__((ext_vector_type(8)))  float  v8f;

#define NUM_NODES 1000000
#define IN_DIM 128
#define HID 256
#define B_EDGE 4096
#define N2 (3 * B_EDGE)      // 12288
#define FAN 10
#define N1 (N2 * FAN)        // 122880

// ---------- scalar bf16 helpers ----------
__device__ __forceinline__ u16 f2b(float f) {
    union { float f; unsigned u; } x; x.f = f;
    unsigned r = x.u + 0x7FFFu + ((x.u >> 16) & 1u);   // round-to-nearest-even
    return (u16)(r >> 16);
}
__device__ __forceinline__ float b2f(u16 h) {
    union { unsigned u; float f; } x; x.u = ((unsigned)h) << 16;
    return x.f;
}
__device__ __forceinline__ unsigned pack2(float a, float b) {
    return (unsigned)f2b(a) | ((unsigned)f2b(b) << 16);
}
__device__ __forceinline__ void unpack8(uint4 q, float* s) {
    s[0] = b2f((u16)(q.x & 0xFFFF)); s[1] = b2f((u16)(q.x >> 16));
    s[2] = b2f((u16)(q.y & 0xFFFF)); s[3] = b2f((u16)(q.y >> 16));
    s[4] = b2f((u16)(q.z & 0xFFFF)); s[5] = b2f((u16)(q.z >> 16));
    s[6] = b2f((u16)(q.w & 0xFFFF)); s[7] = b2f((u16)(q.w >> 16));
}

// ---------- weight convert + transpose: Wt[n][kOff+k] = W[k][n] ----------
__global__ void convT_kernel(const float* __restrict__ W, u16* __restrict__ Wt,
                             int K, int N, int ldK, int kOff) {
    int idx = blockIdx.x * 256 + threadIdx.x;
    if (idx >= K * N) return;
    int k = idx / N, n = idx - k * N;
    Wt[(size_t)n * ldK + kOff + k] = f2b(W[idx]);
}

// ---------- layer-0 gather + mean-aggregate; emits bf16 A0 = [x_dst | agg] ----------
__global__ __launch_bounds__(128) void gather_agg0_kernel(
    const float* __restrict__ nf, const int* __restrict__ gids,
    const int* __restrict__ nidx, u16* __restrict__ A0) {
    int lane = threadIdx.x & 31;
    int row  = blockIdx.x * 4 + (threadIdx.x >> 5);   // < N1
    // dst feature (128 f32 = 1 float4/lane on wave32)
    const float4* dp = (const float4*)(nf + (size_t)gids[row] * IN_DIM);
    float4 d = dp[lane];
    float4 s = make_float4(0.f, 0.f, 0.f, 0.f);
#pragma unroll
    for (int j = 0; j < FAN; ++j) {
        int n = nidx[row * FAN + j];
        const float4* p = (const float4*)(nf + (size_t)gids[n] * IN_DIM);
        float4 v = p[lane];
        s.x += v.x; s.y += v.y; s.z += v.z; s.w += v.w;
    }
    const float inv = 1.0f / FAN;
    u16* o = A0 + (size_t)row * (2 * IN_DIM);
    ushort4 pd; pd.x = f2b(d.x); pd.y = f2b(d.y); pd.z = f2b(d.z); pd.w = f2b(d.w);
    ((ushort4*)o)[lane] = pd;
    ushort4 pa; pa.x = f2b(s.x*inv); pa.y = f2b(s.y*inv); pa.z = f2b(s.z*inv); pa.w = f2b(s.w*inv);
    ((ushort4*)(o + IN_DIM))[lane] = pa;
}

// ---------- layer-1 input build: A1 = [h_dst | mean(h_neigh)] (bf16) ----------
__global__ __launch_bounds__(128) void build_a1_kernel(
    const u16* __restrict__ H1, const int* __restrict__ nidx, u16* __restrict__ A1) {
    int lane = threadIdx.x & 31;
    int row  = blockIdx.x * 4 + (threadIdx.x >> 5);   // < N2
    u16* o = A1 + (size_t)row * (2 * HID);
    // dst row: 256 bf16 = 1 uint4 (8 bf16) per lane — pass through
    uint4 d = ((const uint4*)(H1 + (size_t)row * HID))[lane];
    ((uint4*)o)[lane] = d;
    float s[8] = {0.f,0.f,0.f,0.f,0.f,0.f,0.f,0.f};
#pragma unroll
    for (int j = 0; j < FAN; ++j) {
        int n = nidx[row * FAN + j];
        uint4 q = ((const uint4*)(H1 + (size_t)n * HID))[lane];
        float t[8]; unpack8(q, t);
#pragma unroll
        for (int e = 0; e < 8; ++e) s[e] += t[e];
    }
    const float inv = 1.0f / FAN;
    uint4 pk;
    pk.x = pack2(s[0]*inv, s[1]*inv); pk.y = pack2(s[2]*inv, s[3]*inv);
    pk.z = pack2(s[4]*inv, s[5]*inv); pk.w = pack2(s[6]*inv, s[7]*inv);
    ((uint4*)(o + HID))[lane] = pk;
}

// ---------- z = concat(src,src) * concat(pos,neg)  (bf16) ----------
__global__ __launch_bounds__(128) void zbuild_kernel(const u16* __restrict__ H2,
                                                     u16* __restrict__ Z) {
    int lane = threadIdx.x & 31;
    int r = blockIdx.x * 4 + (threadIdx.x >> 5);      // < 2*B_EDGE
    int srow = r & (B_EDGE - 1);
    int orow = (r < B_EDGE) ? (B_EDGE + srow) : (2 * B_EDGE + srow);
    uint4 qa = ((const uint4*)(H2 + (size_t)srow * HID))[lane];
    uint4 qb = ((const uint4*)(H2 + (size_t)orow * HID))[lane];
    float a[8], b[8]; unpack8(qa, a); unpack8(qb, b);
    uint4 pk;
    pk.x = pack2(a[0]*b[0], a[1]*b[1]); pk.y = pack2(a[2]*b[2], a[3]*b[3]);
    pk.z = pack2(a[4]*b[4], a[5]*b[5]); pk.w = pack2(a[6]*b[6], a[7]*b[7]);
    ((uint4*)(Z + (size_t)r * HID))[lane] = pk;
}

// ---------- final projection: scores = Z2 @ Wp3 + bp3 (f32 out) ----------
__global__ __launch_bounds__(128) void final_proj_kernel(const u16* __restrict__ Z2,
                                                         const float* __restrict__ Wp3,
                                                         const float* __restrict__ bp3,
                                                         float* __restrict__ out) {
    int lane = threadIdx.x & 31;
    int r = blockIdx.x * 4 + (threadIdx.x >> 5);      // < 2*B_EDGE
    uint4 q = ((const uint4*)(Z2 + (size_t)r * HID))[lane];
    float z[8]; unpack8(q, z);
    const float4* w4 = (const float4*)(Wp3 + lane * 8);
    float4 w0 = w4[0], w1 = w4[1];
    float s = z[0]*w0.x + z[1]*w0.y + z[2]*w0.z + z[3]*w0.w
            + z[4]*w1.x + z[5]*w1.y + z[6]*w1.z + z[7]*w1.w;
#pragma unroll
    for (int off = 16; off > 0; off >>= 1) s += __shfl_xor(s, off, 32);
    if (lane == 0) out[r] = s + bp3[0];
}

// ---------- WMMA bf16 GEMM: Out = opt_relu(A[MxK] @ Wt^T + bias), Out bf16 ----------
// A row-major bf16 (K contiguous); Wt is N-major/K-contiguous bf16 (i.e. W transposed).
#define BM 128
#define BN 64
#define BK 32
union Frag16 { uint4 u[2]; v16bf v; };

__global__ __launch_bounds__(256) void gemm_bf16_wmma_kernel(
    const u16* __restrict__ A, const u16* __restrict__ Wt,
    const float* __restrict__ bias, u16* __restrict__ Out,
    int M, int N, int K, int doRelu) {
    __shared__ u16 sA[BM * BK];   // 8 KB
    __shared__ u16 sB[BN * BK];   // 4 KB, N-major / K-contiguous
    const int t     = threadIdx.x;
    const int lane  = t & 31;
    const int wave  = t >> 5;     // 0..7
    const int wm    = wave >> 1;  // 0..3 : wave tile row (32 rows each)
    const int wn    = wave & 1;   // 0..1 : wave tile col (32 cols each)
    const int bm    = blockIdx.y * BM;
    const int bn    = blockIdx.x * BN;
    const int lhalf = (lane >> 4) & 1;  // lanes 16-31 hold the upper K-halves
    const int lrow  = lane & 15;

    v8f acc[2][2];
#pragma unroll
    for (int mi = 0; mi < 2; ++mi)
#pragma unroll
        for (int ni = 0; ni < 2; ++ni)
#pragma unroll
            for (int e = 0; e < 8; ++e) acc[mi][ni][e] = 0.f;

    for (int kt = 0; kt < K; kt += BK) {
        // cooperative A tile load: 128x32 bf16 = 512 uint4, 2 per thread
        {
            int id = t;
            int r = id >> 2, s = (id & 3) * 8;
            *(uint4*)&sA[r * BK + s] = *(const uint4*)&A[(size_t)(bm + r) * K + kt + s];
            id = t + 256; r = id >> 2; s = (id & 3) * 8;
            *(uint4*)&sA[r * BK + s] = *(const uint4*)&A[(size_t)(bm + r) * K + kt + s];
        }
        // B tile (N-major): 64x32 bf16 = 256 uint4, 1 per thread
        {
            int r = t >> 2, s = (t & 3) * 8;
            *(uint4*)&sB[r * BK + s] = *(const uint4*)&Wt[(size_t)(bn + r) * K + kt + s];
        }
        __syncthreads();

        // A fragment (16-bit A 16x32): elem e -> K = (e/8)*16 + lhalf*8 + (e%8), M = lrow
        Frag16 af[2], bf[2];
#pragma unroll
        for (int mi = 0; mi < 2; ++mi) {
            int row = wm * 32 + mi * 16 + lrow;
            int klo = lhalf << 3;
            af[mi].u[0] = *(const uint4*)&sA[row * BK + klo];
            af[mi].u[1] = *(const uint4*)&sA[row * BK + 16 + klo];
        }
        // B fragment (16-bit B 32x16): elem e -> K = (e/8)*16 + lhalf*8 + (e%8), N = lrow
#pragma unroll
        for (int ni = 0; ni < 2; ++ni) {
            int col = wn * 32 + ni * 16 + lrow;
            int kb  = lhalf << 3;
            bf[ni].u[0] = *(const uint4*)&sB[col * BK + kb];
            bf[ni].u[1] = *(const uint4*)&sB[col * BK + kb + 16];
        }
#pragma unroll
        for (int mi = 0; mi < 2; ++mi)
#pragma unroll
            for (int ni = 0; ni < 2; ++ni)
                acc[mi][ni] = __builtin_amdgcn_wmma_f32_16x16x32_bf16(
                    false, af[mi].v, false, bf[ni].v,
                    (short)0, acc[mi][ni], false, false);
        __syncthreads();
    }

    // epilogue: C/D f32 16x16 layout -> VGPR v: M = v + lhalf*8, N = lrow
#pragma unroll
    for (int mi = 0; mi < 2; ++mi)
#pragma unroll
        for (int ni = 0; ni < 2; ++ni)
#pragma unroll
            for (int v = 0; v < 8; ++v) {
                int row = bm + wm * 32 + mi * 16 + lhalf * 8 + v;
                int col = bn + wn * 32 + ni * 16 + lrow;
                float x = acc[mi][ni][v] + bias[col];
                if (doRelu) x = fmaxf(x, 0.f);
                Out[(size_t)row * N + col] = f2b(x);
            }
}

// ---------------------------------------------------------------------------
extern "C" void kernel_launch(void* const* d_in, const int* in_sizes, int n_in,
                              void* d_out, int out_size, void* d_ws, size_t ws_size,
                              hipStream_t stream) {
    const float* node_feat = (const float*)d_in[0];
    const int*   gids0     = (const int*)d_in[1];
    const int*   nidx0     = (const int*)d_in[2];
    const int*   nidx1     = (const int*)d_in[3];
    const float* W_self0   = (const float*)d_in[4];
    const float* W_neigh0  = (const float*)d_in[5];
    const float* b0        = (const float*)d_in[6];
    const float* W_self1   = (const float*)d_in[7];
    const float* W_neigh1  = (const float*)d_in[8];
    const float* b1        = (const float*)d_in[9];
    const float* Wp1       = (const float*)d_in[10];
    const float* bp1       = (const float*)d_in[11];
    const float* Wp2       = (const float*)d_in[12];
    const float* bp2       = (const float*)d_in[13];
    const float* Wp3       = (const float*)d_in[14];
    const float* bp3       = (const float*)d_in[15];
    float* scores = (float*)d_out;

    // -------- workspace partition (bf16 buffers) --------
    char* ws = (char*)d_ws;
    size_t off = 0;
    auto alloc = [&](size_t bytes) -> char* {
        char* p = ws + off;
        off += (bytes + 255) & ~(size_t)255;
        return p;
    };
    u16* Wt0  = (u16*)alloc((size_t)HID * (2 * IN_DIM) * 2);   // [256 x 256] K=256
    u16* Wt1  = (u16*)alloc((size_t)HID * (2 * HID) * 2);      // [256 x 512] K=512
    u16* Wtp1 = (u16*)alloc((size_t)HID * HID * 2);
    u16* Wtp2 = (u16*)alloc((size_t)HID * HID * 2);
    u16* A0   = (u16*)alloc((size_t)N1 * (2 * IN_DIM) * 2);    // [N1 x 256]
    u16* H1   = (u16*)alloc((size_t)N1 * HID * 2);             // [N1 x 256]
    u16* A1   = (u16*)alloc((size_t)N2 * (2 * HID) * 2);       // [N2 x 512]
    u16* H2   = (u16*)alloc((size_t)N2 * HID * 2);             // [N2 x 256]
    u16* Z    = (u16*)alloc((size_t)2 * B_EDGE * HID * 2);     // [8192 x 256]
    u16* Z1   = (u16*)alloc((size_t)2 * B_EDGE * HID * 2);
    u16* Z2   = (u16*)alloc((size_t)2 * B_EDGE * HID * 2);
    (void)ws_size; (void)in_sizes; (void)n_in; (void)out_size;

    // -------- 1) weight convert + transpose (K-concat self||neigh) --------
    auto cvt = [&](const float* W, u16* Wt, int K, int N, int ldK, int kOff) {
        int n = K * N;
        convT_kernel<<<(n + 255) / 256, 256, 0, stream>>>(W, Wt, K, N, ldK, kOff);
    };
    cvt(W_self0,  Wt0,  IN_DIM, HID, 2 * IN_DIM, 0);
    cvt(W_neigh0, Wt0,  IN_DIM, HID, 2 * IN_DIM, IN_DIM);
    cvt(W_self1,  Wt1,  HID,    HID, 2 * HID,    0);
    cvt(W_neigh1, Wt1,  HID,    HID, 2 * HID,    HID);
    cvt(Wp1,      Wtp1, HID,    HID, HID,        0);
    cvt(Wp2,      Wtp2, HID,    HID, HID,        0);

    // -------- 2) layer-0 gather + aggregate --------
    gather_agg0_kernel<<<N1 / 4, 128, 0, stream>>>(node_feat, gids0, nidx0, A0);

    // -------- 3) layer-0 GEMM: H1 = relu(A0 @ [Ws0;Wn0] + b0) --------
    gemm_bf16_wmma_kernel<<<dim3(HID / BN, N1 / BM), 256, 0, stream>>>(
        A0, Wt0, b0, H1, N1, HID, 2 * IN_DIM, 1);

    // -------- 4) layer-1 input build --------
    build_a1_kernel<<<N2 / 4, 128, 0, stream>>>(H1, nidx1, A1);

    // -------- 5) layer-1 GEMM: H2 = A1 @ [Ws1;Wn1] + b1 (no relu) --------
    gemm_bf16_wmma_kernel<<<dim3(HID / BN, N2 / BM), 256, 0, stream>>>(
        A1, Wt1, b1, H2, N2, HID, 2 * HID, 0);

    // -------- 6) z = concat(src,src) * concat(pos,neg) --------
    zbuild_kernel<<<(2 * B_EDGE) / 4, 128, 0, stream>>>(H2, Z);

    // -------- 7) predictor GEMMs --------
    gemm_bf16_wmma_kernel<<<dim3(HID / BN, (2 * B_EDGE) / BM), 256, 0, stream>>>(
        Z, Wtp1, bp1, Z1, 2 * B_EDGE, HID, HID, 1);
    gemm_bf16_wmma_kernel<<<dim3(HID / BN, (2 * B_EDGE) / BM), 256, 0, stream>>>(
        Z1, Wtp2, bp2, Z2, 2 * B_EDGE, HID, HID, 1);

    // -------- 8) final projection to scores (f32) --------
    final_proj_kernel<<<(2 * B_EDGE) / 4, 128, 0, stream>>>(Z2, Wp3, bp3, scores);
}
